// M2TREC_72129680769646
// MI455X (gfx1250) — matrix operand order repaired and checked by reading, not compiled, not run on hardware
//
#include <hip/hip_runtime.h>
#include <hip/hip_bf16.h>
#include <cstdint>
#include <cstddef>

// Model constants (match reference)
#define B_     1024
#define L_     50
#define D_     128
#define H_     4
#define DH_    32
#define FF_    512
#define T_     52
#define DEPTH_ 2

typedef __attribute__((ext_vector_type(16))) _Float16 v16h;
typedef __attribute__((ext_vector_type(8)))  _Float16 v8h;
typedef __attribute__((ext_vector_type(8)))  float    v8f;

// ---------------------------------------------------------------------------
// Device helpers
// ---------------------------------------------------------------------------
__device__ __forceinline__ float gelu_tanh(float x) {
  // jax.nn.gelu default (approximate=True)
  float x3 = x * x * x;
  return 0.5f * x * (1.0f + tanhf(0.7978845608028654f * (x + 0.044715f * x3)));
}

// A fragment: 16(M) x 32(K) f16.  ISA 7.12.2: lane l holds row m=l&15;
// element j -> k = ((j&8)<<1) + ((l>>4)<<3) + (j&7)  => two contiguous 8-half groups.
__device__ __forceinline__ v16h load_afrag(const _Float16* A, int row, int K, int kk, int lane) {
  const _Float16* p = A + (size_t)row * K + kk + ((lane >> 4) << 3);
  v8h lo = *(const v8h*)(p);
  v8h hi = *(const v8h*)(p + 16);
  v16h r;
#pragma unroll
  for (int i = 0; i < 8; ++i) { r[i] = lo[i]; r[8 + i] = hi[i]; }
  return r;
}
// Same fragment but converting from fp32 source rows.
__device__ __forceinline__ v16h load_afrag(const float* A, int row, int K, int kk, int lane) {
  const float* p = A + (size_t)row * K + kk + ((lane >> 4) << 3);
  v16h r;
#pragma unroll
  for (int i = 0; i < 8; ++i) { r[i] = (_Float16)p[i]; r[8 + i] = (_Float16)p[16 + i]; }
  return r;
}
// B fragment: 32(K) x 16(N) f16.  lane l holds column n=l&15;
// element j -> k = ((j&8)<<1) + ((l>>4)<<3) + (j&7)  => two contiguous 8-half groups of transposed weight Wt[N][K].
__device__ __forceinline__ v16h load_bfrag(const _Float16* Wt, int ncol, int K, int kk, int lane) {
  const _Float16* p = Wt + (size_t)ncol * K + kk + ((lane >> 4) << 3);
  v8h lo = *(const v8h*)(p);
  v8h hi = *(const v8h*)(p + 16);
  v16h r;
#pragma unroll
  for (int i = 0; i < 8; ++i) { r[i] = lo[i]; r[8 + i] = hi[i]; }
  return r;
}

// ---------------------------------------------------------------------------
// Generic WMMA GEMM:  C[M,N] = A[M,K] @ Wt[N,K]^T (+bias) (+act) (+residual)
// Register-blocked: each wave32 owns a (16*MBLK) x (16*NBLK) C macro-tile.
// Per K-step: MBLK A-fragment loads + NBLK B-fragment loads feed MBLK*NBLK
// independent v_wmma_f32_16x16x32_f16 accumulator chains (1.5 b128 loads/wmma
// at 2x4 blocking vs 4 for the naive 1x1 tile).
// OUTMODE: 0 = f16 store, 1 = fp32 residual add (out32 += v), 2 = fp32 store.
// ACT:     0 = none, 1 = gelu(tanh)
// ---------------------------------------------------------------------------
template <typename AT, int OUTMODE, int ACT, int MBLK, int NBLK>
__global__ void k_gemm(const AT* __restrict__ A, const _Float16* __restrict__ Wt,
                       const float* __restrict__ bias,
                       _Float16* __restrict__ out16, float* __restrict__ out32,
                       int M, int N, int K) {
  int wave = blockIdx.x * (blockDim.x >> 5) + (threadIdx.x >> 5);
  int lane = threadIdx.x & 31;
  int tilesN = N / (16 * NBLK);
  int totTiles = (M / (16 * MBLK)) * tilesN;
  if (wave >= totTiles) return;              // whole-wave uniform exit: EXEC stays all-ones for WMMA
  int tm = (wave / tilesN) * (16 * MBLK);
  int tn = (wave % tilesN) * (16 * NBLK);

  // For residual-add epilogues, prefetch the RMW destination rows now so the
  // read half of "+=" overlaps the K-loop (global_prefetch_b8, write hint).
  if (OUTMODE == 1) {
#pragma unroll
    for (int mb = 0; mb < MBLK; ++mb) {
      __builtin_prefetch(out32 + (size_t)(tm + mb * 16 + (lane & 15)) * N + tn, 1, 3);
    }
  }

  v8f c[MBLK][NBLK];
#pragma unroll
  for (int mb = 0; mb < MBLK; ++mb)
#pragma unroll
    for (int nb = 0; nb < NBLK; ++nb)
#pragma unroll
      for (int i = 0; i < 8; ++i) c[mb][nb][i] = 0.f;

  for (int kk = 0; kk < K; kk += 32) {
    v16h a[MBLK];
#pragma unroll
    for (int mb = 0; mb < MBLK; ++mb)
      a[mb] = load_afrag(A, tm + mb * 16 + (lane & 15), K, kk, lane);
#pragma unroll
    for (int nb = 0; nb < NBLK; ++nb) {
      v16h b = load_bfrag(Wt, tn + nb * 16 + (lane & 15), K, kk, lane);
#pragma unroll
      for (int mb = 0; mb < MBLK; ++mb)
        c[mb][nb] = __builtin_amdgcn_wmma_f32_16x16x32_f16(false, a[mb], false, b,
                                                           (short)0, c[mb][nb], false, false);
    }
  }

#pragma unroll
  for (int nb = 0; nb < NBLK; ++nb) {
    int ncol = tn + nb * 16 + (lane & 15);
    float bv = bias ? bias[ncol] : 0.0f;
#pragma unroll
    for (int mb = 0; mb < MBLK; ++mb) {
      int mbase = tm + mb * 16 + ((lane >> 4) << 3);   // C/D layout: VGPR i -> M = i (+8 for hi lanes)
#pragma unroll
      for (int i = 0; i < 8; ++i) {
        float v = c[mb][nb][i] + bv;
        if (ACT == 1) v = gelu_tanh(v);
        size_t idx = (size_t)(mbase + i) * N + ncol;
        if (OUTMODE == 0)      out16[idx] = (_Float16)v;
        else if (OUTMODE == 1) out32[idx] += v;
        else                   out32[idx] = v;
      }
    }
  }
}

// ---------------------------------------------------------------------------
// Embedding gather: one row (b,l) per block, 128 threads.
// ---------------------------------------------------------------------------
__global__ void k_gather(const int* __restrict__ ids, const float* __restrict__ emb,
                         float* __restrict__ out) {
  int r = blockIdx.x, tid = threadIdx.x;
  out[(size_t)r * D_ + tid] = emb[(size_t)ids[r] * D_ + tid];
}

// ---------------------------------------------------------------------------
// Build h = LN(tok + pos_emb[pos_idx]).  tok = CLS / 0.5*(hs+ha) / MASK.
// Prefix-valid mask => pos = min(t, len) for t<T-1, len+1 for MASK slot.
// ---------------------------------------------------------------------------
__global__ void k_build_h(const float* __restrict__ hs, const float* __restrict__ ha,
                          const float* __restrict__ cls, const float* __restrict__ mtok,
                          const float* __restrict__ pos_emb, const unsigned char* __restrict__ mask,
                          const float* __restrict__ lnw, const float* __restrict__ lnb,
                          float* __restrict__ h) {
  __shared__ float red[D_];
  __shared__ int s_len;
  int bt = blockIdx.x, b = bt / T_, t = bt % T_, tid = threadIdx.x;
  if (tid == 0) {
    int c = 0;
    for (int j = 0; j < L_; ++j) c += (mask[b * L_ + j] != 0);
    s_len = c;
  }
  __syncthreads();
  int len = s_len;
  float v;
  if (t == 0)            v = cls[tid];
  else if (t == T_ - 1)  v = mtok[tid];
  else                   v = 0.5f * (hs[((size_t)b * L_ + (t - 1)) * D_ + tid] +
                                     ha[((size_t)b * L_ + (t - 1)) * D_ + tid]);
  int pos = (t == T_ - 1) ? (len + 1) : (t < len ? t : len);
  v += pos_emb[(size_t)pos * D_ + tid];
  // LayerNorm over D=128
  red[tid] = v; __syncthreads();
  for (int s = D_ / 2; s > 0; s >>= 1) { if (tid < s) red[tid] += red[tid + s]; __syncthreads(); }
  float mean = red[0] / D_; __syncthreads();
  float d = v - mean;
  red[tid] = d * d; __syncthreads();
  for (int s = D_ / 2; s > 0; s >>= 1) { if (tid < s) red[tid] += red[tid + s]; __syncthreads(); }
  float var = red[0] / D_;
  h[(size_t)bt * D_ + tid] = d * rsqrtf(var + 1e-5f) * lnw[tid] + lnb[tid];
}

// ---------------------------------------------------------------------------
// LayerNorm of h rows -> f16 (GEMM input) or f32.
// ---------------------------------------------------------------------------
template <typename OT>
__global__ void k_layernorm(const float* __restrict__ x, const float* __restrict__ w,
                            const float* __restrict__ b, OT* __restrict__ y) {
  __shared__ float red[D_];
  int row = blockIdx.x, tid = threadIdx.x;
  float v = x[(size_t)row * D_ + tid];
  red[tid] = v; __syncthreads();
  for (int s = D_ / 2; s > 0; s >>= 1) { if (tid < s) red[tid] += red[tid + s]; __syncthreads(); }
  float mean = red[0] / D_; __syncthreads();
  float d = v - mean;
  red[tid] = d * d; __syncthreads();
  for (int s = D_ / 2; s > 0; s >>= 1) { if (tid < s) red[tid] += red[tid + s]; __syncthreads(); }
  float var = red[0] / D_;
  y[(size_t)row * D_ + tid] = (OT)(d * rsqrtf(var + 1e-5f) * w[tid] + b[tid]);
}

// ---------------------------------------------------------------------------
// Weight convert+transpose: W[K,N] f32 -> Wt[N,K] f16 (WMMA-B-friendly).
// ---------------------------------------------------------------------------
__global__ void k_w2h_t(const float* __restrict__ W, _Float16* __restrict__ Wt, int K, int N) {
  int i = blockIdx.x * blockDim.x + threadIdx.x;
  if (i >= K * N) return;
  int k = i / N, n = i % N;
  Wt[(size_t)n * K + k] = (_Float16)W[i];
}

// ---------------------------------------------------------------------------
// Self-attention: one (b,h) per block.  K/V staged in LDS; one query row per
// thread; two-pass online softmax (causal + padding mask).  q,k,v come from
// the fused QKV GEMM output (f16, layout [b*T+t, {q:0,k:128,v:256} + h*32+d]).
// ---------------------------------------------------------------------------
__global__ void k_attn(const _Float16* __restrict__ qkv, const unsigned char* __restrict__ mask,
                       _Float16* __restrict__ o16) {
  __shared__ _Float16 sk[T_ * DH_];
  __shared__ _Float16 sv[T_ * DH_];
  int bh = blockIdx.x, b = bh / H_, h = bh % H_;
  for (int i = threadIdx.x; i < T_ * DH_; i += blockDim.x) {
    int t = i / DH_, d = i % DH_;
    size_t base = (size_t)(b * T_ + t) * (3 * D_);
    sk[i] = qkv[base + D_     + h * DH_ + d];
    sv[i] = qkv[base + 2 * D_ + h * DH_ + d];
  }
  __syncthreads();
  int t = threadIdx.x;
  if (t >= T_) return;

  float q[DH_];
  size_t qb = (size_t)(b * T_ + t) * (3 * D_) + h * DH_;
#pragma unroll
  for (int d = 0; d < DH_; ++d) q[d] = (float)qkv[qb + d];

  const float scale = 0.17677669529663687f;  // 1/sqrt(32)
  // pass 1: running max over valid (causal + padding) keys
  float mx = -3.4e38f;
  for (int j = 0; j <= t; ++j) {
    bool valid = (j == 0) || (j == T_ - 1) || (mask[b * L_ + (j - 1)] != 0);
    if (!valid) continue;
    float acc = 0.f;
#pragma unroll
    for (int d = 0; d < DH_; ++d) acc += q[d] * (float)sk[j * DH_ + d];
    mx = fmaxf(mx, acc * scale);
  }
  // pass 2: accumulate softmax numerator into o
  float den = 0.f;
  float o[DH_];
#pragma unroll
  for (int d = 0; d < DH_; ++d) o[d] = 0.f;
  for (int j = 0; j <= t; ++j) {
    bool valid = (j == 0) || (j == T_ - 1) || (mask[b * L_ + (j - 1)] != 0);
    if (!valid) continue;
    float acc = 0.f;
#pragma unroll
    for (int d = 0; d < DH_; ++d) acc += q[d] * (float)sk[j * DH_ + d];
    float e = __expf(acc * scale - mx);
    den += e;
#pragma unroll
    for (int d = 0; d < DH_; ++d) o[d] += e * (float)sv[j * DH_ + d];
  }
  float inv = 1.0f / den;
  size_t ob = (size_t)(b * T_ + t) * D_ + h * DH_;
#pragma unroll
  for (int d = 0; d < DH_; ++d) o16[ob + d] = (_Float16)(o[d] * inv);
}

// ---------------------------------------------------------------------------
// Cross attention: one (b,t) per block (128 threads).
// sim[h][j] = q[b,t,h,:] . k[b,j,:]   (no scaling, per reference)
// out[b,t,d] = sum_j (sum_h softmax_j sim[h][j]) * hv[b,j,d]
// Writes directly into the sliced output regions.
// ---------------------------------------------------------------------------
__global__ void k_cross(const _Float16* __restrict__ q16, const float* __restrict__ kbuf,
                        const float* __restrict__ hv, const unsigned char* __restrict__ mask,
                        float* __restrict__ out0, float* __restrict__ out1) {
  int bt = blockIdx.x, b = bt / T_, t = bt % T_;
  if (t == 0) return;  // g[:,0] is dropped by the reference slicing
  __shared__ float qf[D_];
  __shared__ float sim[H_][L_];
  __shared__ float wsum[L_];
  int tid = threadIdx.x;
  qf[tid] = (float)q16[(size_t)bt * D_ + tid];
  __syncthreads();

  for (int p = tid; p < H_ * L_; p += D_) {
    int h = p / L_, j = p % L_;
    float acc;
    if (mask[b * L_ + j]) {
      const float* kp = kbuf + ((size_t)b * L_ + j) * DH_;
      acc = 0.f;
#pragma unroll
      for (int d = 0; d < DH_; ++d) acc += qf[h * DH_ + d] * kp[d];
    } else {
      acc = -3.4e38f;
    }
    sim[h][j] = acc;
  }
  __syncthreads();
  if (tid < H_) {
    int h = tid;
    float mx = -3.4e38f;
    for (int j = 0; j < L_; ++j) mx = fmaxf(mx, sim[h][j]);
    float den = 0.f;
    for (int j = 0; j < L_; ++j) {
      float e = (sim[h][j] <= -3.0e38f) ? 0.f : __expf(sim[h][j] - mx);
      sim[h][j] = e;
      den += e;
    }
    float inv = 1.0f / den;
    for (int j = 0; j < L_; ++j) sim[h][j] *= inv;
  }
  __syncthreads();
  if (tid < L_) wsum[tid] = sim[0][tid] + sim[1][tid] + sim[2][tid] + sim[3][tid];
  __syncthreads();

  float acc = 0.f;
  const float* vb = hv + (size_t)b * L_ * D_ + tid;
  for (int j = 0; j < L_; ++j) acc += wsum[j] * vb[(size_t)j * D_];

  if (t < T_ - 1) out0[((size_t)b * L_ + (t - 1)) * D_ + tid] = acc;
  else            out1[(size_t)b * D_ + tid] = acc;
}

// ---------------------------------------------------------------------------
// Host orchestration
// ---------------------------------------------------------------------------
extern "C" void kernel_launch(void* const* d_in, const int* in_sizes, int n_in,
                              void* d_out, int out_size, void* d_ws, size_t ws_size,
                              hipStream_t stream) {
  (void)in_sizes; (void)n_in; (void)out_size; (void)ws_size;

  const int*           seq_ids  = (const int*)d_in[0];
  const int*           attr_ids = (const int*)d_in[1];
  const unsigned char* mask     = (const unsigned char*)d_in[2];  // jnp bool_ = 1 byte
  const float* emb_seq  = (const float*)d_in[3];
  const float* emb_attr = (const float*)d_in[4];
  const float* pos_emb  = (const float*)d_in[5];
  const float* cls_tok  = (const float*)d_in[6];
  const float* mask_tok = (const float*)d_in[7];
  const float* ln_w     = (const float*)d_in[8];
  const float* ln_b     = (const float*)d_in[9];
  const float* tq_seq   = (const float*)d_in[10];
  const float* tq_attr  = (const float*)d_in[11];
  const float* tk_seq   = (const float*)d_in[12];
  const float* tk_attr  = (const float*)d_in[13];
  const float* t_qkv    = (const float*)d_in[14];
  const float* t_out    = (const float*)d_in[15];
  const float* t_ln1w   = (const float*)d_in[16];
  const float* t_ln1b   = (const float*)d_in[17];
  const float* t_ln2w   = (const float*)d_in[18];
  const float* t_ln2b   = (const float*)d_in[19];
  const float* t_ff1w   = (const float*)d_in[20];
  const float* t_ff1b   = (const float*)d_in[21];
  const float* t_ff2w   = (const float*)d_in[22];
  const float* t_ff2b   = (const float*)d_in[23];
  const float* t_lnfw   = (const float*)d_in[24];
  const float* t_lnfb   = (const float*)d_in[25];

  const int M  = B_ * T_;   // 53248 rows through the encoder (mult of 32)
  const int M2 = B_ * L_;   // 51200 rows for cross-attn keys  (mult of 32)

  // -------- workspace carve (256B aligned chunks) --------
  char* wp = (char*)d_ws;
  auto carve = [&](size_t bytes) -> char* {
    char* p = wp;
    wp += (bytes + 255) & ~(size_t)255;
    return p;
  };
  float*     h     = (float*)carve((size_t)M * D_ * sizeof(float));
  float*     hs    = (float*)carve((size_t)M2 * D_ * sizeof(float));
  float*     ha    = (float*)carve((size_t)M2 * D_ * sizeof(float));
  _Float16*  hn16  = (_Float16*)carve((size_t)M * D_ * sizeof(_Float16));
  _Float16*  qkv16 = (_Float16*)carve((size_t)M * 3 * D_ * sizeof(_Float16));
  _Float16*  o16   = (_Float16*)carve((size_t)M * D_ * sizeof(_Float16));   // also reused as cross q
  _Float16*  ff16  = (_Float16*)carve((size_t)M * FF_ * sizeof(_Float16));
  float*     kbuf  = (float*)carve((size_t)M2 * DH_ * sizeof(float));
  _Float16*  wqkv16  = (_Float16*)carve((size_t)DEPTH_ * D_ * 3 * D_ * sizeof(_Float16));
  _Float16*  wout16  = (_Float16*)carve((size_t)DEPTH_ * D_ * D_ * sizeof(_Float16));
  _Float16*  wff1_16 = (_Float16*)carve((size_t)DEPTH_ * D_ * FF_ * sizeof(_Float16));
  _Float16*  wff2_16 = (_Float16*)carve((size_t)DEPTH_ * FF_ * D_ * sizeof(_Float16));
  _Float16*  wtqs16  = (_Float16*)carve((size_t)D_ * D_ * sizeof(_Float16));
  _Float16*  wtqa16  = (_Float16*)carve((size_t)D_ * D_ * sizeof(_Float16));
  _Float16*  wtks16  = (_Float16*)carve((size_t)D_ * DH_ * sizeof(_Float16));
  _Float16*  wtka16  = (_Float16*)carve((size_t)D_ * DH_ * sizeof(_Float16));

  // -------- convert + transpose all GEMM weights to f16 [N,K] --------
  auto conv = [&](const float* W, _Float16* Wt, int K, int N) {
    int n = K * N;
    k_w2h_t<<<(n + 255) / 256, 256, 0, stream>>>(W, Wt, K, N);
  };
  for (int l = 0; l < DEPTH_; ++l) {
    conv(t_qkv  + (size_t)l * D_ * 3 * D_, wqkv16  + (size_t)l * D_ * 3 * D_, D_, 3 * D_);
    conv(t_out  + (size_t)l * D_ * D_,     wout16  + (size_t)l * D_ * D_,     D_, D_);
    conv(t_ff1w + (size_t)l * D_ * FF_,    wff1_16 + (size_t)l * D_ * FF_,    D_, FF_);
    conv(t_ff2w + (size_t)l * FF_ * D_,    wff2_16 + (size_t)l * FF_ * D_,    FF_, D_);
  }
  conv(tq_seq,  wtqs16, D_, D_);
  conv(tq_attr, wtqa16, D_, D_);
  conv(tk_seq,  wtks16, D_, DH_);
  conv(tk_attr, wtka16, D_, DH_);

  // -------- embeddings + input build --------
  k_gather<<<M2, D_, 0, stream>>>(seq_ids, emb_seq, hs);
  k_gather<<<M2, D_, 0, stream>>>(attr_ids, emb_attr, ha);
  k_build_h<<<B_ * T_, D_, 0, stream>>>(hs, ha, cls_tok, mask_tok, pos_emb, mask, ln_w, ln_b, h);

  const int WPB = 8;  // 8 waves (macro-tiles) per 256-thread block
  auto gblocks = [&](int m, int n, int mblk, int nblk) {
    int tiles = (m / (16 * mblk)) * (n / (16 * nblk));
    return (tiles + WPB - 1) / WPB;
  };

  // -------- transformer layers --------
  for (int l = 0; l < DEPTH_; ++l) {
    k_layernorm<_Float16><<<M, D_, 0, stream>>>(h, t_ln1w + l * D_, t_ln1b + l * D_, hn16);
    k_gemm<_Float16, 0, 0, 2, 4><<<gblocks(M, 3 * D_, 2, 4), 256, 0, stream>>>(
        hn16, wqkv16 + (size_t)l * D_ * 3 * D_, nullptr, qkv16, nullptr, M, 3 * D_, D_);
    k_attn<<<B_ * H_, 64, 0, stream>>>(qkv16, mask, o16);
    k_gemm<_Float16, 1, 0, 2, 4><<<gblocks(M, D_, 2, 4), 256, 0, stream>>>(
        o16, wout16 + (size_t)l * D_ * D_, nullptr, nullptr, h, M, D_, D_);
    k_layernorm<_Float16><<<M, D_, 0, stream>>>(h, t_ln2w + l * D_, t_ln2b + l * D_, hn16);
    k_gemm<_Float16, 0, 1, 2, 4><<<gblocks(M, FF_, 2, 4), 256, 0, stream>>>(
        hn16, wff1_16 + (size_t)l * D_ * FF_, t_ff1b + l * FF_, ff16, nullptr, M, FF_, D_);
    k_gemm<_Float16, 1, 0, 2, 4><<<gblocks(M, D_, 2, 4), 256, 0, stream>>>(
        ff16, wff2_16 + (size_t)l * FF_ * D_, t_ff2b + l * D_, nullptr, h, M, D_, FF_);
  }

  // -------- final LN --------
  k_layernorm<_Float16><<<M, D_, 0, stream>>>(h, t_lnfw, t_lnfb, hn16);

  // -------- cross attention heads, writing directly to sliced outputs --------
  float* O = (float*)d_out;
  const size_t P0 = (size_t)B_ * L_ * D_;
  const size_t P1 = (size_t)B_ * D_;

  // seq head
  k_gemm<_Float16, 0, 0, 2, 4><<<gblocks(M, D_, 2, 4), 256, 0, stream>>>(
      hn16, wtqs16, nullptr, o16, nullptr, M, D_, D_);
  k_gemm<float, 2, 0, 2, 2><<<gblocks(M2, DH_, 2, 2), 256, 0, stream>>>(
      hs, wtks16, nullptr, nullptr, kbuf, M2, DH_, D_);
  k_cross<<<B_ * T_, D_, 0, stream>>>(o16, kbuf, hs, mask, O, O + P0);

  // attr head
  k_gemm<_Float16, 0, 0, 2, 4><<<gblocks(M, D_, 2, 4), 256, 0, stream>>>(
      hn16, wtqa16, nullptr, o16, nullptr, M, D_, D_);
  k_gemm<float, 2, 0, 2, 2><<<gblocks(M2, DH_, 2, 2), 256, 0, stream>>>(
      ha, wtka16, nullptr, nullptr, kbuf, M2, DH_, D_);
  k_cross<<<B_ * T_, D_, 0, stream>>>(o16, kbuf, ha, mask, O + P0 + P1, O + P0 + P1 + P0);
}
